// GAT_67070209294885
// MI455X (gfx1250) — matrix operand order repaired and checked by reading, not compiled, not run on hardware
//
#include <hip/hip_runtime.h>

typedef __attribute__((ext_vector_type(16))) _Float16 v16h;
typedef __attribute__((ext_vector_type(8)))  _Float16 v8h;
typedef __attribute__((ext_vector_type(8)))  float    v8f;

#define SLOPE 0.2f

__device__ __forceinline__ unsigned fmap(float x) {
  unsigned u = __float_as_uint(x);
  return (u & 0x80000000u) ? ~u : (u | 0x80000000u);
}
__device__ __forceinline__ float funmap(unsigned u) {
  return (u & 0x80000000u) ? __uint_as_float(u & 0x7FFFFFFFu) : __uint_as_float(~u);
}
__device__ __forceinline__ float lrelu(float v) { return v > 0.f ? v : SLOPE * v; }
__device__ __forceinline__ float eluf(float v)  { return v > 0.f ? v : (__expf(v) - 1.f); }

// ---------------- zero accumulators ----------------
__global__ void zero_ws_k(float* p, size_t n) {
  size_t i = (size_t)blockIdx.x * blockDim.x + threadIdx.x;
  if (i < n) p[i] = 0.f;
}

// ---------------- GEMM1: h1pre[N,64] = x[N,256] @ W1[256,64] via WMMA f16 ----------------
// Block: 128 threads (4 waves). Block tile: 32 rows x 64 cols. Wave w owns cols
// [16w,16w+16) and both 16-row halves (two accumulators -> 2 WMMAs per K-step).
__global__ __launch_bounds__(128)
void gemm1_wmma(const float* __restrict__ x, const float* __restrict__ W1,
                float* __restrict__ h1, int Nn) {
  __shared__ __align__(32) _Float16 As[32 * 32];   // row-major [m][k]
  __shared__ __align__(32) _Float16 Bst[64 * 32];  // col-major [n][k]
  const int tid  = threadIdx.x;
  const int wave = tid >> 5;
  const int lane = tid & 31;
  const int m0 = blockIdx.x * 32;
  const int n0 = wave * 16;
  v8f acc0 = {};
  v8f acc1 = {};

  for (int kk = 0; kk < 256; kk += 32) {
    // stage A tile 32x32 (f32 -> f16): thread -> 8 consecutive floats, one 16B LDS store
    {
      int r = tid >> 2;            // 0..31
      int c = (tid & 3) * 8;       // 0,8,16,24
      int row = m0 + r; if (row >= Nn) row = Nn - 1;
      const float* p = x + (size_t)row * 256 + kk + c;
      float4 v0 = *(const float4*)(p);
      float4 v1 = *(const float4*)(p + 4);
      v8h hv;
      hv[0] = (_Float16)v0.x; hv[1] = (_Float16)v0.y;
      hv[2] = (_Float16)v0.z; hv[3] = (_Float16)v0.w;
      hv[4] = (_Float16)v1.x; hv[5] = (_Float16)v1.y;
      hv[6] = (_Float16)v1.z; hv[7] = (_Float16)v1.w;
      *(v8h*)(As + r * 32 + c) = hv;
    }
    // stage B tile 32x64 into col-major Bst; 128 threads x 4 float4
    {
      int kr = tid >> 2;          // 0..31
      int cb = (tid & 3) * 16;    // 0,16,32,48
#pragma unroll
      for (int j = 0; j < 4; ++j) {
        float4 v = *(const float4*)(W1 + (size_t)(kk + kr) * 64 + cb + j * 4);
        Bst[(cb + j * 4 + 0) * 32 + kr] = (_Float16)v.x;
        Bst[(cb + j * 4 + 1) * 32 + kr] = (_Float16)v.y;
        Bst[(cb + j * 4 + 2) * 32 + kr] = (_Float16)v.z;
        Bst[(cb + j * 4 + 3) * 32 + kr] = (_Float16)v.w;
      }
    }
    __syncthreads();
    // B fragment: lane<16 -> col n, K 0..15 ; lane>=16 -> col n, K 16..31
    int ncol = n0 + (lane & 15);
    int koff = (lane >> 4) * 16;
    v16h b = *(const v16h*)(Bst + ncol * 32 + koff);
    // A fragments (match B): lane<16 -> K 0..15 ; lane>=16 -> K 16..31
    int m = lane & 15;
    int khalf = (lane >> 4) * 16;
    {
      v8h alo = *(const v8h*)(As + m * 32 + khalf);
      v8h ahi = *(const v8h*)(As + m * 32 + 8 + khalf);
      v16h a;
#pragma unroll
      for (int i = 0; i < 8; ++i) { a[i] = alo[i]; a[8 + i] = ahi[i]; }
      acc0 = __builtin_amdgcn_wmma_f32_16x16x32_f16(false, a, false, b, (short)0, acc0,
                                                    false, false);
    }
    {
      v8h alo = *(const v8h*)(As + (16 + m) * 32 + khalf);
      v8h ahi = *(const v8h*)(As + (16 + m) * 32 + 8 + khalf);
      v16h a;
#pragma unroll
      for (int i = 0; i < 8; ++i) { a[i] = alo[i]; a[8 + i] = ahi[i]; }
      acc1 = __builtin_amdgcn_wmma_f32_16x16x32_f16(false, a, false, b, (short)0, acc1,
                                                    false, false);
    }
    __syncthreads();
  }
  // C/D layout: lane<16: vgpr r -> (M=r, N=lane); lane>=16: (M=8+r, N=lane-16)
  int n = n0 + (lane & 15);
  int mb = (lane >> 4) * 8;
  float* base = h1 + (size_t)(m0 + mb) * 64 + n;
  if (m0 + 31 < Nn) {   // fast path: whole 32-row tile in range (N%32==0 case)
#pragma unroll
    for (int r = 0; r < 8; ++r) base[(size_t)r * 64] = acc0[r];
#pragma unroll
    for (int r = 0; r < 8; ++r) base[(size_t)(16 + r) * 64] = acc1[r];
  } else {
#pragma unroll
    for (int r = 0; r < 8; ++r) {
      if (m0 + mb + r      < Nn) base[(size_t)r * 64]        = acc0[r];
      if (m0 + mb + r + 16 < Nn) base[(size_t)(16 + r) * 64] = acc1[r];
    }
  }
}

// ---------------- layer-1 attention coefficients per node ----------------
__global__ void attn1_proj(const float* __restrict__ h1, const float* __restrict__ att_s,
                           const float* __restrict__ att_d, float* __restrict__ asrc,
                           float* __restrict__ adst, int Nn) {
  int t = blockIdx.x * blockDim.x + threadIdx.x;  // n*8 + h
  if (t >= Nn * 8) return;
  int n = t >> 3, h = t & 7;
  const float* hp = h1 + (size_t)n * 64 + h * 8;
  float s = 0.f, d = 0.f;
#pragma unroll
  for (int c = 0; c < 8; ++c) { float v = hp[c]; s += v * att_s[h * 8 + c]; d += v * att_d[h * 8 + c]; }
  asrc[t] = s; adst[t] = d;
}

// ---------------- layer-1 edge phases ----------------
__global__ void edge1_max(const int* __restrict__ ei, const float* __restrict__ asrc,
                          const float* __restrict__ adst, unsigned* __restrict__ maxb,
                          int E, int EP) {
  int t = blockIdx.x * blockDim.x + threadIdx.x;
  if (t >= EP * 8) return;
  int e = t >> 3, h = t & 7;
  int s, d;
  if (e < E) { s = ei[e]; d = ei[E + e]; } else { s = d = e - E; }
  float v = lrelu(asrc[(size_t)s * 8 + h] + adst[(size_t)d * 8 + h]);
  atomicMax(maxb + (size_t)d * 8 + h, fmap(v));
}

__global__ void edge1_exp(const int* __restrict__ ei, const float* __restrict__ asrc,
                          const float* __restrict__ adst, const unsigned* __restrict__ maxb,
                          float* __restrict__ den, float* __restrict__ ex, int E, int EP) {
  int t = blockIdx.x * blockDim.x + threadIdx.x;
  if (t >= EP * 8) return;
  int e = t >> 3, h = t & 7;
  int s, d;
  if (e < E) { s = ei[e]; d = ei[E + e]; } else { s = d = e - E; }
  float v = lrelu(asrc[(size_t)s * 8 + h] + adst[(size_t)d * 8 + h]);
  float m = funmap(maxb[(size_t)d * 8 + h]);
  float xv = __expf(v - m);
  ex[t] = xv;
  atomicAdd(den + (size_t)d * 8 + h, xv);
}

__global__ void edge1_agg(const int* __restrict__ ei, const float* __restrict__ ex,
                          const float* __restrict__ den, const float* __restrict__ h1,
                          float* __restrict__ alpha_out, float* __restrict__ agg,
                          int E, int EP) {
  int t = blockIdx.x * blockDim.x + threadIdx.x;
  if (t >= EP * 8) return;
  int e = t >> 3, h = t & 7;
  int s, d;
  if (e < E) { s = ei[e]; d = ei[E + e]; } else { s = d = e - E; }
  float a = ex[t] / (den[(size_t)d * 8 + h] + 1e-16f);
  alpha_out[t] = a;
  const float* hp = h1 + (size_t)s * 64 + h * 8;
  float* ap = agg + (size_t)d * 64 + h * 8;
#pragma unroll
  for (int c = 0; c < 8; ++c) atomicAdd(ap + c, a * hp[c]);
}

// ---------------- finish layer1 (bias+ELU), project to layer2, attn coeffs ----------------
__global__ __launch_bounds__(128)
void layer2_proj(const float* __restrict__ agg1, const float* __restrict__ b1,
                 const float* __restrict__ W2, const float* __restrict__ att_s2,
                 const float* __restrict__ att_d2, float* __restrict__ hp2,
                 float* __restrict__ asrc2, float* __restrict__ adst2, int Nn) {
  int n = blockIdx.x * blockDim.x + threadIdx.x;
  if (n >= Nn) return;
  float t[64];
#pragma unroll
  for (int c = 0; c < 64; ++c) t[c] = eluf(agg1[(size_t)n * 64 + c] + b1[c]);
  float s = 0.f, d = 0.f;
#pragma unroll
  for (int c2 = 0; c2 < 10; ++c2) {
    float acc = 0.f;
#pragma unroll
    for (int c = 0; c < 64; ++c) acc += t[c] * W2[c * 10 + c2];
    hp2[(size_t)n * 10 + c2] = acc;
    s += acc * att_s2[c2];
    d += acc * att_d2[c2];
  }
  asrc2[n] = s; adst2[n] = d;
}

// ---------------- layer-2 edge phases (1 head, 10 ch) ----------------
__global__ void edge2_max(const int* __restrict__ ei, const float* __restrict__ asrc,
                          const float* __restrict__ adst, unsigned* __restrict__ maxb,
                          int E, int EP) {
  int e = blockIdx.x * blockDim.x + threadIdx.x;
  if (e >= EP) return;
  int s, d;
  if (e < E) { s = ei[e]; d = ei[E + e]; } else { s = d = e - E; }
  float v = lrelu(asrc[s] + adst[d]);
  atomicMax(maxb + d, fmap(v));
}

__global__ void edge2_exp(const int* __restrict__ ei, const float* __restrict__ asrc,
                          const float* __restrict__ adst, const unsigned* __restrict__ maxb,
                          float* __restrict__ den, float* __restrict__ ex, int E, int EP) {
  int e = blockIdx.x * blockDim.x + threadIdx.x;
  if (e >= EP) return;
  int s, d;
  if (e < E) { s = ei[e]; d = ei[E + e]; } else { s = d = e - E; }
  float v = lrelu(asrc[s] + adst[d]);
  float xv = __expf(v - funmap(maxb[d]));
  ex[e] = xv;
  atomicAdd(den + d, xv);
}

__global__ void edge2_agg(const int* __restrict__ ei, const float* __restrict__ ex,
                          const float* __restrict__ den, const float* __restrict__ hp2,
                          float* __restrict__ agg2, int E, int EP) {
  int e = blockIdx.x * blockDim.x + threadIdx.x;
  if (e >= EP) return;
  int s, d;
  if (e < E) { s = ei[e]; d = ei[E + e]; } else { s = d = e - E; }
  float a = ex[e] / (den[d] + 1e-16f);
  const float* hp = hp2 + (size_t)s * 10;
  float* ap = agg2 + (size_t)d * 10;
#pragma unroll
  for (int c = 0; c < 10; ++c) atomicAdd(ap + c, a * hp[c]);
}

// ---------------- final: ELU(agg2+b2) @ Wm + bm, ELU ----------------
__global__ __launch_bounds__(128)
void final_out(const float* __restrict__ agg2, const float* __restrict__ b2,
               const float* __restrict__ Wm, const float* __restrict__ bm,
               float* __restrict__ out, int Nn) {
  int n = blockIdx.x;
  int j = threadIdx.x;
  __shared__ float h2[10];
  if (j < 10) h2[j] = eluf(agg2[(size_t)n * 10 + j] + b2[j]);
  __syncthreads();
  float acc = bm[j];
#pragma unroll
  for (int c = 0; c < 10; ++c) acc += h2[c] * Wm[c * 128 + j];
  out[(size_t)n * 128 + j] = eluf(acc);
}

extern "C" void kernel_launch(void* const* d_in, const int* in_sizes, int n_in,
                              void* d_out, int out_size, void* d_ws, size_t ws_size,
                              hipStream_t stream) {
  const float* x   = (const float*)d_in[0];
  const int*   ei  = (const int*)d_in[1];
  const float* W1  = (const float*)d_in[2];
  const float* as1 = (const float*)d_in[3];
  const float* ad1 = (const float*)d_in[4];
  const float* b1  = (const float*)d_in[5];
  const float* W2  = (const float*)d_in[6];
  const float* as2 = (const float*)d_in[7];
  const float* ad2 = (const float*)d_in[8];
  const float* b2  = (const float*)d_in[9];
  const float* Wm  = (const float*)d_in[10];
  const float* bm  = (const float*)d_in[11];
  float* out = (float*)d_out;

  const int Nn = in_sizes[0] / 256;   // 100000
  const int E  = in_sizes[1] / 2;     // 1600000
  const int EP = E + Nn;              // with self-loops

  float* ws = (float*)d_ws;
  size_t o = 0;
  float* h1pre = ws + o; o += (size_t)Nn * 64;
  float* asrc1 = ws + o; o += (size_t)Nn * 8;
  float* adst1 = ws + o; o += (size_t)Nn * 8;
  float* ex1   = ws + o; o += (size_t)EP * 8;
  float* hp2   = ws + o; o += (size_t)Nn * 10;
  float* asrc2 = ws + o; o += (size_t)Nn;
  float* adst2 = ws + o; o += (size_t)Nn;
  float* ex2   = ws + o; o += (size_t)EP;
  size_t zoff = o;                                 // zeroed region begins here
  unsigned* max1 = (unsigned*)(ws + o); o += (size_t)Nn * 8;
  float* den1  = ws + o; o += (size_t)Nn * 8;
  float* agg1  = ws + o; o += (size_t)Nn * 64;
  unsigned* max2 = (unsigned*)(ws + o); o += (size_t)Nn;
  float* den2  = ws + o; o += (size_t)Nn;
  float* agg2  = ws + o; o += (size_t)Nn * 10;
  size_t zcount = o - zoff;

  float* alpha_out = out + (size_t)Nn * 128;  // output tuple: [out | alpha]

  // 1. zero accumulators (fresh every call -> deterministic under graph replay)
  zero_ws_k<<<(unsigned)((zcount + 255) / 256), 256, 0, stream>>>(ws + zoff, zcount);
  // 2. WMMA GEMM: h1pre = x @ W1
  gemm1_wmma<<<(Nn + 31) / 32, 128, 0, stream>>>(x, W1, h1pre, Nn);
  // 3. attention coefficients layer 1
  attn1_proj<<<(Nn * 8 + 255) / 256, 256, 0, stream>>>(h1pre, as1, ad1, asrc1, adst1, Nn);
  // 4-6. layer-1 segment softmax + aggregation
  int g1 = (EP * 8 + 255) / 256;
  edge1_max<<<g1, 256, 0, stream>>>(ei, asrc1, adst1, max1, E, EP);
  edge1_exp<<<g1, 256, 0, stream>>>(ei, asrc1, adst1, max1, den1, ex1, E, EP);
  edge1_agg<<<g1, 256, 0, stream>>>(ei, ex1, den1, h1pre, alpha_out, agg1, E, EP);
  // 7. ELU + projection to layer-2 feature space + attn coeffs
  layer2_proj<<<(Nn + 127) / 128, 128, 0, stream>>>(agg1, b1, W2, as2, ad2, hp2, asrc2,
                                                    adst2, Nn);
  // 8-10. layer-2 segment softmax + aggregation
  int g2 = (EP + 255) / 256;
  edge2_max<<<g2, 256, 0, stream>>>(ei, asrc2, adst2, max2, E, EP);
  edge2_exp<<<g2, 256, 0, stream>>>(ei, asrc2, adst2, max2, den2, ex2, E, EP);
  edge2_agg<<<g2, 256, 0, stream>>>(ei, ex2, den2, hp2, agg2, E, EP);
  // 11. final dense layer + ELU
  final_out<<<Nn, 128, 0, stream>>>(agg2, b2, Wm, bm, out, Nn);
}
